// Reformer_2130303779157
// MI455X (gfx1250) — matrix operand order repaired and checked by reading, not compiled, not run on hardware
//
#include <hip/hip_runtime.h>
#include <hip/hip_bf16.h>
#include <math.h>

// ---------------------------------------------------------------------------
// Reformer forward for MI455X (gfx1250, wave32, WMMA).
// All dense GEMMs and the bucketed-attention matmuls run on
// v_wmma_f32_16x16x32_bf16 (bf16 inputs converted on the fly, f32 accum).
// LDS tiles are stored so that every WMMA fragment read is contiguous
// (ds_load_b128) and rows are padded to avoid bank conflicts.
// ---------------------------------------------------------------------------

typedef __bf16 bf16_t;
typedef bf16_t v16bf __attribute__((ext_vector_type(16)));
typedef bf16_t v4bf  __attribute__((ext_vector_type(4)));
typedef float  v8f   __attribute__((ext_vector_type(8)));

#define BSZ    4
#define NSEQ   2048
#define FDIM   1024
#define DEPTHL 2
#define NHEAD  8
#define BUCKET 64
#define NHASH  4
#define DH     128
#define NBUCK  16
#define BH     32          // BSZ*NHEAD
#define MROWS  8192        // BSZ*NSEQ
#define NC     128         // NHASH*NSEQ/BUCKET
#define NHN    8192        // NHASH*NSEQ

#define APAD   40          // 32 K + 8 pad  (80B row stride, conflict-free frag reads)
#define QPAD   136         // 128 + 8 pad   (272B row stride)
#define DPAD   132         // dots row pad (f32)

__device__ __forceinline__ bf16_t f2bf(float f) {
  unsigned u = __builtin_bit_cast(unsigned, f);
  unsigned r = (u + 0x7FFFu + ((u >> 16) & 1u)) >> 16;
  unsigned short s = (unsigned short)r;
  return __builtin_bit_cast(bf16_t, s);
}

__device__ __forceinline__ v4bf pack4(const float* p) {
  v4bf pk;
  pk[0] = f2bf(p[0]); pk[1] = f2bf(p[1]); pk[2] = f2bf(p[2]); pk[3] = f2bf(p[3]);
  return pk;
}

// ---------------------------------------------------------------------------
// Embedding: x = tok_emb[inp] + pos_emb ; split into x0 (cols 0..F) / x1.
// ---------------------------------------------------------------------------
__global__ void embed_kernel(const float* __restrict__ te, const float* __restrict__ pe,
                             const int* __restrict__ inp,
                             float* __restrict__ x0, float* __restrict__ x1) {
  size_t idx = (size_t)blockIdx.x * 256 + threadIdx.x;   // MROWS*FDIM threads
  int f = (int)(idx & (FDIM - 1));
  int m = (int)(idx >> 10);
  int n = m & (NSEQ - 1);
  int tok = inp[m];
  x0[idx] = te[(size_t)tok * 2048 + f]        + pe[(size_t)n * 2048 + f];
  x1[idx] = te[(size_t)tok * 2048 + FDIM + f] + pe[(size_t)n * 2048 + FDIM + f];
}

// ---------------------------------------------------------------------------
// Row LayerNorm (one 256-thread workgroup per row, generic feature dim).
// ---------------------------------------------------------------------------
__global__ __launch_bounds__(256) void ln_kernel(const float* __restrict__ x,
                                                 const float* __restrict__ g,
                                                 const float* __restrict__ bb,
                                                 float* __restrict__ y, int Fd) {
  int row = blockIdx.x, tid = threadIdx.x;
  const float* xr = x + (size_t)row * Fd;
  __shared__ float red[256];
  float s = 0.f;
  for (int f = tid; f < Fd; f += 256) s += xr[f];
  red[tid] = s; __syncthreads();
  for (int o = 128; o > 0; o >>= 1) { if (tid < o) red[tid] += red[tid + o]; __syncthreads(); }
  float mu = red[0] / (float)Fd;
  __syncthreads();
  float s2 = 0.f;
  for (int f = tid; f < Fd; f += 256) { float dd = xr[f] - mu; s2 += dd * dd; }
  red[tid] = s2; __syncthreads();
  for (int o = 128; o > 0; o >>= 1) { if (tid < o) red[tid] += red[tid + o]; __syncthreads(); }
  float rstd = rsqrtf(red[0] / (float)Fd + 1e-5f);
  for (int f = tid; f < Fd; f += 256)
    y[(size_t)row * Fd + f] = (xr[f] - mu) * rstd * g[f] + bb[f];
}

// ---------------------------------------------------------------------------
// Generic WMMA GEMM:  C = epilogue(A[M,K] @ B[K,N] + bias (+res))
// 64x64 tile per 128-thread block (4 waves, 16 M-rows each, 4 N-subtiles).
// A tile stored [m][k], B tile stored TRANSPOSED [n][k] so both fragment
// reads are contiguous 16-bf16 runs (2x ds_load_b128 each).
// ---------------------------------------------------------------------------
__global__ __launch_bounds__(128) void gemm_bf16(const float* __restrict__ A,
                                                 const float* __restrict__ Bm,
                                                 const float* __restrict__ bias,
                                                 const float* __restrict__ res,
                                                 float* __restrict__ C,
                                                 int M, int K, int Nn, int gelu_act) {
  __shared__ alignas(16) bf16_t As[64][APAD];    // [m][k]
  __shared__ alignas(16) bf16_t BsT[64][APAD];   // [n][k]  (transposed B tile)
  const int tid  = threadIdx.x;
  const int wave = tid >> 5;
  const int lane = tid & 31;
  const int tM = blockIdx.y * 64;
  const int tN = blockIdx.x * 64;
  v8f acc[4] = {};
  for (int k0 = 0; k0 < K; k0 += 32) {
    // stage A: 64x32, 4 consecutive floats per thread -> global_load_b128 + ds_store_b64
    for (int q = tid; q < (64 * 32) / 4; q += 128) {
      int e4 = q * 4;
      int r = e4 >> 5, cc = e4 & 31;
      *(v4bf*)&As[r][cc] = pack4(&A[(size_t)(tM + r) * K + (k0 + cc)]);
    }
    // stage B transposed: read 4 consecutive N-columns, scatter into [n][k]
    for (int q = tid; q < (32 * 64) / 4; q += 128) {
      int e4 = q * 4;
      int r = e4 >> 6, cc = e4 & 63;
      const float* gp = &Bm[(size_t)(k0 + r) * Nn + (tN + cc)];
      BsT[cc + 0][r] = f2bf(gp[0]);
      BsT[cc + 1][r] = f2bf(gp[1]);
      BsT[cc + 2][r] = f2bf(gp[2]);
      BsT[cc + 3][r] = f2bf(gp[3]);
    }
    __syncthreads();
    // A fragment: 16x32 bf16 (lanes 0-15: K 0-15; 16-31: K 16-31)
    v16bf af;
    const int arow = wave * 16 + (lane & 15);
    const int asel = (lane < 16) ? 0 : 16;
#pragma unroll
    for (int e = 0; e < 16; ++e) af[e] = As[arow][asel + e];
    const int bsel = (lane < 16) ? 0 : 16;
    const int cn   = lane & 15;
#pragma unroll
    for (int nt = 0; nt < 4; ++nt) {
      v16bf bfr;
#pragma unroll
      for (int e = 0; e < 16; ++e) bfr[e] = BsT[nt * 16 + cn][bsel + e];
      acc[nt] = __builtin_amdgcn_wmma_f32_16x16x32_bf16(false, af, false, bfr,
                                                        (short)0, acc[nt], false, false);
    }
    __syncthreads();
  }
  const int rh = (lane < 16) ? 0 : 8;
  const int cn = lane & 15;
#pragma unroll
  for (int nt = 0; nt < 4; ++nt)
#pragma unroll
    for (int r = 0; r < 8; ++r) {
      int grow = tM + wave * 16 + rh + r;
      int gcol = tN + nt * 16 + cn;
      float vv = acc[nt][r];
      if (bias) vv += bias[gcol];
      if (gelu_act) vv = 0.5f * vv * (1.0f + erff(vv * 0.7071067811865475f));  // exact gelu
      if (res) vv += res[(size_t)grow * Nn + gcol];
      C[(size_t)grow * Nn + gcol] = vv;
    }
}

// ---------------------------------------------------------------------------
// LSH rotation + argmax -> bucket key in [0,64). One thread per (bh,n,hash).
// ---------------------------------------------------------------------------
__global__ void bucket_kernel(const float* __restrict__ qk, const float* __restrict__ rot,
                              int* __restrict__ buckets) {
  int idx = blockIdx.x * 256 + threadIdx.x;            // BH*NSEQ*NHASH
  int r  = idx & 3;
  int n  = (idx >> 2) & (NSEQ - 1);
  int bh = idx >> 13;
  int b = bh >> 3, head = bh & 7;
  const float* q = qk + ((size_t)(b * NSEQ + n) * NHEAD + head) * DH;
  float sv[8];
  for (int j = 0; j < 8; ++j) {
    float s = 0.f;
    for (int dh = 0; dh < DH; ++dh) s += q[dh] * rot[dh * 32 + r * 8 + j];
    sv[j] = s;
  }
  // argmax over [s0..s7, -s0..-s7], first-max tie rule like jnp.argmax
  float best = sv[0]; int bj = 0;
  for (int j = 1; j < 8; ++j) if (sv[j]  > best) { best = sv[j];  bj = j; }
  for (int j = 0; j < 8; ++j) if (-sv[j] > best) { best = -sv[j]; bj = 8 + j; }
  buckets[(size_t)bh * NHN + r * NSEQ + n] = bj + r * NBUCK;
}

// ---------------------------------------------------------------------------
// Stable sort replicating argsort(buckets*N + ticker):
// counting sort over 64 bucket values, enumerating elements in (ticker,hash)
// order so ties resolve exactly as JAX's stable argsort. 64 threads = 1 bucket each.
// ---------------------------------------------------------------------------
__global__ __launch_bounds__(64) void sort_kernel(const int* __restrict__ buckets,
                                                  int* __restrict__ sidx) {
  int bh = blockIdx.x, tb = threadIdx.x;
  __shared__ int cnt[64];
  const int* bk = buckets + (size_t)bh * NHN;
  int c = 0;
  for (int j = 0; j < NHN; ++j) {
    int i = ((j & 3) << 11) + (j >> 2);                // i = hash*NSEQ + ticker
    if (bk[i] == tb) ++c;
  }
  cnt[tb] = c;
  __syncthreads();
  if (tb == 0) { int a = 0; for (int q = 0; q < 64; ++q) { int t = cnt[q]; cnt[q] = a; a += t; } }
  __syncthreads();
  int pos = cnt[tb];
  for (int j = 0; j < NHN; ++j) {
    int i = ((j & 3) << 11) + (j >> 2);
    if (bk[i] == tb) sidx[(size_t)bh * NHN + (pos++)] = i;
  }
}

// ---------------------------------------------------------------------------
// Bucketed attention. One block per (chunk, bh): 64 queries vs 128 kv
// (current + previous chunk), both matmuls on WMMA.
//   Aq  [64][QPAD]  bf16 : queries, later reused for softmax probabilities P
//   region2 (34.8KB), time-multiplexed:
//     Bk  [128][QPAD] bf16 : normalized keys [kv][dh]  (dots B-frag contiguous)
//     dots[64][DPAD]  f32  : raw scores
//     BvT [128][QPAD] bf16 : values TRANSPOSED [dh][kv] (out B-frag contiguous)
// Outputs scattered directly to (bh, hash, token) so no unperm is needed.
// ---------------------------------------------------------------------------
__global__ __launch_bounds__(128) void attn_kernel(const float* __restrict__ qk,
                                                   const float* __restrict__ vsrc,
                                                   const int* __restrict__ sidx,
                                                   float* __restrict__ o_all,
                                                   float* __restrict__ lg) {
  __shared__ alignas(16) bf16_t Aq[64][QPAD];                 // 17408 B
  __shared__ alignas(16) char   reg2[128 * QPAD * 2];         // 34816 B
  __shared__ int kvi[128];
  __shared__ int kvt[128];
  bf16_t (*Bk)[QPAD]  = (bf16_t(*)[QPAD])reg2;
  float  (*dots)[DPAD] = (float(*)[DPAD])reg2;
  bf16_t (*BvT)[QPAD] = (bf16_t(*)[QPAD])reg2;

  const int c  = blockIdx.x;
  const int bh = blockIdx.y;
  const int b = bh >> 3, head = bh & 7;
  const int tid = threadIdx.x;
  const int wave = tid >> 5, lane = tid & 31;
  const int cprev = (c + NC - 1) & (NC - 1);

  {   // kv index/token caches (rows 0..63 = this chunk = queries, 64..127 = prev chunk)
    int j = tid;
    int chunk = (j < 64) ? c : cprev;
    int i = sidx[(size_t)bh * NHN + chunk * 64 + (j & 63)];
    kvi[j] = i;
    kvt[j] = i & (NSEQ - 1);
  }
  __syncthreads();
  {   // queries -> Aq (two threads per row, vectorized 4-wide)
    int r = tid >> 1, hsel = tid & 1;
    const float* src = qk + ((size_t)(b * NSEQ + kvt[r]) * NHEAD + head) * DH + hsel * 64;
    bf16_t* dst = &Aq[r][hsel * 64];
    for (int e = 0; e < 64; e += 4) *(v4bf*)(dst + e) = pack4(src + e);
  }
  {   // normalized keys -> Bk (one thread per kv row)
    int r = tid;
    const float* src = qk + ((size_t)(b * NSEQ + kvt[r]) * NHEAD + head) * DH;
    float ssq = 1e-12f;
    for (int e = 0; e < DH; ++e) { float xv = src[e]; ssq += xv * xv; }
    float inv = rsqrtf(ssq);
    for (int e = 0; e < DH; e += 4) {
      v4bf pk;
      pk[0] = f2bf(src[e] * inv);     pk[1] = f2bf(src[e + 1] * inv);
      pk[2] = f2bf(src[e + 2] * inv); pk[3] = f2bf(src[e + 3] * inv);
      *(v4bf*)&Bk[r][e] = pk;
    }
  }
  __syncthreads();

  // dots[64x128] = bq @ bk2^T  (contraction over DH; Bk rows are dh-contiguous)
  v8f accd[8] = {};
  {
    const int arow = wave * 16 + (lane & 15);
    const int asel = (lane < 16) ? 0 : 16;
    const int bsel = (lane < 16) ? 0 : 16;
    const int cn = lane & 15;
    for (int ks = 0; ks < 4; ++ks) {
      v16bf af;
#pragma unroll
      for (int e = 0; e < 16; ++e) af[e] = Aq[arow][ks * 32 + asel + e];
#pragma unroll
      for (int nt = 0; nt < 8; ++nt) {
        v16bf bfr;
#pragma unroll
        for (int e = 0; e < 16; ++e) bfr[e] = Bk[nt * 16 + cn][ks * 32 + bsel + e];
        accd[nt] = __builtin_amdgcn_wmma_f32_16x16x32_bf16(false, af, false, bfr,
                                                           (short)0, accd[nt], false, false);
      }
    }
  }
  __syncthreads();                                    // all Bk reads done
  {   // scale + self-mask, spill scores to LDS (overwrites Bk)
    const int rh = (lane < 16) ? 0 : 8;
    const int cn = lane & 15;
#pragma unroll
    for (int nt = 0; nt < 8; ++nt)
#pragma unroll
      for (int r = 0; r < 8; ++r) {
        int row = wave * 16 + rh + r;
        int col = nt * 16 + cn;
        float dv = accd[nt][r] * 0.08838834764831845f;   // DH^-0.5
        if (kvt[col] == kvt[row]) dv = -50000.0f;        // SELF_ATTN_VALUE
        dots[row][col] = dv;
      }
  }
  __syncthreads();
  if (tid < 64) {   // row softmax, p -> Aq (bq dead), lse -> lg scatter
    int row = tid;
    float mx = -3.4e38f;
    for (int cc = 0; cc < 128; ++cc) mx = fmaxf(mx, dots[row][cc]);
    float ssum = 0.f;
    for (int cc = 0; cc < 128; ++cc) ssum += expf(dots[row][cc] - mx);
    float lse = mx + logf(ssum);
    for (int cc = 0; cc < 128; cc += 4) {
      v4bf pk;
      pk[0] = f2bf(expf(dots[row][cc + 0] - lse));
      pk[1] = f2bf(expf(dots[row][cc + 1] - lse));
      pk[2] = f2bf(expf(dots[row][cc + 2] - lse));
      pk[3] = f2bf(expf(dots[row][cc + 3] - lse));
      *(v4bf*)&Aq[row][cc] = pk;
    }
    int i = kvi[row];
    lg[((size_t)bh * NHASH + (i >> 11)) * NSEQ + (i & (NSEQ - 1))] = lse;
  }
  __syncthreads();
  {   // values -> BvT transposed [dh][kv]  (overwrites dots)
    int r = tid;
    const float* src = vsrc + ((size_t)(b * NSEQ + kvt[r]) * NHEAD + head) * DH;
    for (int e = 0; e < DH; ++e) BvT[e][r] = f2bf(src[e]);
  }
  __syncthreads();

  // out[64x128] = p @ bv2  (B fragment from BvT rows: kv-contiguous)
  v8f acco[8] = {};
  {
    const int arow = wave * 16 + (lane & 15);
    const int asel = (lane < 16) ? 0 : 16;
    const int bsel = (lane < 16) ? 0 : 16;
    const int cn = lane & 15;
    for (int ks = 0; ks < 4; ++ks) {
      v16bf af;
#pragma unroll
      for (int e = 0; e < 16; ++e) af[e] = Aq[arow][ks * 32 + asel + e];
#pragma unroll
      for (int nt = 0; nt < 8; ++nt) {
        v16bf bfr;
#pragma unroll
        for (int e = 0; e < 16; ++e) bfr[e] = BvT[nt * 16 + cn][ks * 32 + bsel + e];
        acco[nt] = __builtin_amdgcn_wmma_f32_16x16x32_bf16(false, af, false, bfr,
                                                           (short)0, acco[nt], false, false);
      }
    }
  }
  {   // scatter to o_all[bh][hash][token][dh]
    const int rh = (lane < 16) ? 0 : 8;
    const int cn = lane & 15;
#pragma unroll
    for (int nt = 0; nt < 8; ++nt)
#pragma unroll
      for (int r = 0; r < 8; ++r) {
        int row = wave * 16 + rh + r;
        int dh = nt * 16 + cn;
        int i = kvi[row];
        o_all[(((size_t)bh * NHASH + (i >> 11)) * NSEQ + (i & (NSEQ - 1))) * DH + dh] = acco[nt][r];
      }
  }
}

// ---------------------------------------------------------------------------
// Combine hash rounds: softmax over logits across NHASH, weighted sum of o_all,
// write in (B,N,H,DH) layout for the output projection GEMM.
// ---------------------------------------------------------------------------
__global__ void combine_kernel(const float* __restrict__ o_all, const float* __restrict__ lg,
                               float* __restrict__ ao) {
  size_t idx = (size_t)blockIdx.x * 256 + threadIdx.x;  // BH*NSEQ*DH
  int dh = (int)(idx & (DH - 1));
  int t  = (int)((idx >> 7) & (NSEQ - 1));
  int bh = (int)(idx >> 18);
  float l[NHASH];
  float mx = -3.4e38f;
  for (int h = 0; h < NHASH; ++h) {
    l[h] = lg[((size_t)bh * NHASH + h) * NSEQ + t];
    mx = fmaxf(mx, l[h]);
  }
  float sum = 0.f;
  for (int h = 0; h < NHASH; ++h) { l[h] = expf(l[h] - mx); sum += l[h]; }
  float inv = 1.0f / sum;
  float out = 0.f;
  for (int h = 0; h < NHASH; ++h)
    out += o_all[(((size_t)bh * NHASH + h) * NSEQ + t) * DH + dh] * (l[h] * inv);
  int b = bh >> 3, head = bh & 7;
  ao[((size_t)(b * NSEQ + t) * NHEAD + head) * DH + dh] = out;
}

// ---------------------------------------------------------------------------
// concat(x0, x1) along features for the final LN + head.
// ---------------------------------------------------------------------------
__global__ void concat_kernel(const float* __restrict__ a, const float* __restrict__ bsrc,
                              float* __restrict__ y) {
  size_t idx = (size_t)blockIdx.x * 256 + threadIdx.x;  // MROWS*2F
  int f = (int)(idx & (2 * FDIM - 1));
  size_t m = idx >> 11;
  y[idx] = (f < FDIM) ? a[m * FDIM + f] : bsrc[m * FDIM + (f - FDIM)];
}

// ---------------------------------------------------------------------------
extern "C" void kernel_launch(void* const* d_in, const int* in_sizes, int n_in,
                              void* d_out, int out_size, void* d_ws, size_t ws_size,
                              hipStream_t stream) {
  (void)in_sizes; (void)n_in; (void)out_size; (void)ws_size;
  const float* tok_emb  = (const float*)d_in[0];
  const float* pos_emb  = (const float*)d_in[1];
  const float* ln_att_g = (const float*)d_in[2];
  const float* ln_att_b = (const float*)d_in[3];
  const float* wqk      = (const float*)d_in[4];
  const float* wv       = (const float*)d_in[5];
  const float* wo       = (const float*)d_in[6];
  const float* bo       = (const float*)d_in[7];
  const float* rot      = (const float*)d_in[8];
  const float* ln_ff_g  = (const float*)d_in[9];
  const float* ln_ff_b  = (const float*)d_in[10];
  const float* w1       = (const float*)d_in[11];
  const float* b1       = (const float*)d_in[12];
  const float* w2       = (const float*)d_in[13];
  const float* b2       = (const float*)d_in[14];
  const float* out_g    = (const float*)d_in[15];
  const float* out_b    = (const float*)d_in[16];
  const float* ow       = (const float*)d_in[17];
  const float* ob       = (const float*)d_in[18];
  const int*   inp      = (const int*)d_in[19];

  // Workspace carve-up (~372 MB total, assumed available in d_ws).
  float* W = (float*)d_ws;
  size_t off = 0;
  auto take = [&](size_t n) { float* p = W + off; off += n; return p; };
  const size_t AN = (size_t)MROWS * FDIM;
  float* act0 = take(AN);
  float* act1 = take(AN);
  float* act2 = take(AN);
  float* lnb  = take(AN);
  float* qkb  = take(AN);
  float* vb   = take(AN);
  float* aob  = take(AN);
  float* big  = take((size_t)MROWS * 4096);            // o_all / FFN hidden / final concat+LN
  float* lgb  = take((size_t)BH * NHASH * NSEQ);
  int* buckets = (int*)take((size_t)BH * NHN);
  int* sidx    = (int*)take((size_t)BH * NHN);

  embed_kernel<<<(MROWS * FDIM) / 256, 256, 0, stream>>>(tok_emb, pos_emb, inp, act0, act1);

  float* pa = act0;   // x0
  float* pb = act1;   // x1
  float* pc = act2;   // spare

  dim3 gF(FDIM / 64, MROWS / 64);
  dim3 gFF(4096 / 64, MROWS / 64);

  for (int d = 0; d < DEPTHL; ++d) {
    // ---- attention block ----
    ln_kernel<<<MROWS, 256, 0, stream>>>(pb, ln_att_g + d * FDIM, ln_att_b + d * FDIM, lnb, FDIM);
    gemm_bf16<<<gF, 128, 0, stream>>>(lnb, wqk + (size_t)d * FDIM * FDIM, nullptr, nullptr,
                                      qkb, MROWS, FDIM, FDIM, 0);
    gemm_bf16<<<gF, 128, 0, stream>>>(lnb, wv + (size_t)d * FDIM * FDIM, nullptr, nullptr,
                                      vb, MROWS, FDIM, FDIM, 0);
    bucket_kernel<<<(BH * NSEQ * NHASH) / 256, 256, 0, stream>>>(
        qkb, rot + (size_t)d * DH * NHASH * (NBUCK / 2), buckets);
    sort_kernel<<<BH, 64, 0, stream>>>(buckets, sidx);
    attn_kernel<<<dim3(NC, BH), 128, 0, stream>>>(qkb, vb, sidx, big, lgb);
    combine_kernel<<<(BH * NSEQ * DH) / 256, 256, 0, stream>>>(big, lgb, aob);
    gemm_bf16<<<gF, 128, 0, stream>>>(aob, wo + (size_t)d * FDIM * FDIM, bo + d * FDIM,
                                      pa /*residual*/, pc, MROWS, FDIM, FDIM, 0);
    { float* t = pa; pa = pb; pb = pc; pc = t; }

    // ---- FFN block ----
    ln_kernel<<<MROWS, 256, 0, stream>>>(pb, ln_ff_g + d * FDIM, ln_ff_b + d * FDIM, lnb, FDIM);
    gemm_bf16<<<gFF, 128, 0, stream>>>(lnb, w1 + (size_t)d * FDIM * 4096, b1 + (size_t)d * 4096,
                                       nullptr, big, MROWS, FDIM, 4096, 1 /*gelu*/);
    gemm_bf16<<<gF, 128, 0, stream>>>(big, w2 + (size_t)d * 4096 * FDIM, b2 + d * FDIM,
                                      pa /*residual*/, pc, MROWS, 4096, FDIM, 0);
    { float* t = pa; pa = pb; pb = pc; pc = t; }
  }

  // ---- output head ----
  float* xcat = big;                                   // MROWS * 2F
  float* xln  = big + (size_t)MROWS * 2 * FDIM;        // second half of big region
  concat_kernel<<<(MROWS * 2 * FDIM) / 256, 256, 0, stream>>>(pa, pb, xcat);
  ln_kernel<<<MROWS, 256, 0, stream>>>(xcat, out_g, out_b, xln, 2 * FDIM);
  dim3 gO(256 / 64, MROWS / 64);
  gemm_bf16<<<gO, 128, 0, stream>>>(xln, ow, ob, nullptr, (float*)d_out,
                                    MROWS, 2 * FDIM, 256, 0);
}
